// SAUp_38946763440335
// MI455X (gfx1250) — compile-verified, matrix operand order checked by reading, not run on hardware
//
#include <hip/hip_runtime.h>
#include <hip/hip_bf16.h>
#include <math.h>

// ---------------------------------------------------------------------------
// DCNv2 + BN/ReLU + bilinear x2 upsample + 1x1 + sigmoid for MI455X (gfx1250)
// Matrix work on v_wmma_f32_16x16x32_bf16 (wave32, f32 accumulate).
// A-matrices pre-transposed to bf16 [k][m][c] so WMMA A-fragments are direct
// 32-byte global loads per lane (no LDS for A). B tiles double-buffered in
// LDS -> one barrier per K chunk.
// ---------------------------------------------------------------------------

typedef float v8f __attribute__((ext_vector_type(8)));
typedef __bf16 v16bf __attribute__((ext_vector_type(16)));

#define CHI   256
#define CHO   128
#define Hdim  64
#define Wdim  64
#define BATCH 8
#define PLANE (Hdim * Wdim)          // 4096
#define KTAPS 9
#define MPAD_OFF 32                  // offset-conv M padded 27 -> 32

__device__ __forceinline__ __bf16 f2bf(float f) {
  union { float f; unsigned u; } in; in.f = f;
  unsigned u = in.u;
  u += 0x7FFFu + ((u >> 16) & 1u);          // round-to-nearest-even
  union { unsigned short s; __bf16 b; } out;
  out.s = (unsigned short)(u >> 16);
  return out.b;
}

__device__ __forceinline__ float sigmoidf_(float z) {
  return 1.0f / (1.0f + __expf(-z));
}

// ---------------------------------------------------------------------------
// Kernel 0: reorder conv weights to bf16 Wt[k][m][c] (c contiguous) so a WMMA
// A-fragment is one contiguous 32B load per lane.
//   Wt_dcn : [9][128][256]  (294912 elems)
//   Wt_off : [9][ 32][256]  ( 73728 elems, rows m>=27 zero)
// ---------------------------------------------------------------------------
#define WT_DCN_ELEMS (KTAPS * CHO * CHI)        // 294912
#define WT_OFF_ELEMS (KTAPS * MPAD_OFF * CHI)   //  73728

__global__ __launch_bounds__(256) void reorder_weights(
    const float* __restrict__ w_dcn, const float* __restrict__ w_off,
    __bf16* __restrict__ wt_dcn, __bf16* __restrict__ wt_off) {
  const int idx = blockIdx.x * blockDim.x + threadIdx.x;
  if (idx < WT_DCN_ELEMS) {
    const int c = idx & (CHI - 1);
    const int m = (idx >> 8) & (CHO - 1);
    const int k = idx >> 15;
    wt_dcn[idx] = f2bf(w_dcn[(size_t)(m * CHI + c) * KTAPS + k]);
  } else if (idx < WT_DCN_ELEMS + WT_OFF_ELEMS) {
    const int j = idx - WT_DCN_ELEMS;
    const int c = j & (CHI - 1);
    const int m = (j >> 8) & (MPAD_OFF - 1);
    const int k = j >> 13;
    wt_off[j] = (m < 27)
        ? f2bf(w_off[(size_t)(m * CHI + c) * KTAPS + k])
        : (__bf16)0.0f;
  }
}

// ---------------------------------------------------------------------------
// Kernel 1: offset/mask conv.  M=27(pad 32), N=64 per block, K=2304.
// Block = (batch b, row h). 8 waves = 2 M-blocks x 4 N-subtiles.
// ---------------------------------------------------------------------------
__global__ __launch_bounds__(256) void dcn_offset_conv_wmma(
    const float* __restrict__ x, const __bf16* __restrict__ wt_off,
    const float* __restrict__ b_off,
    float* __restrict__ offY, float* __restrict__ offX,
    float* __restrict__ maskW) {
  __shared__ __align__(64) __bf16 ldsB[2][64 * 32];   // [p][kq], double buffer

  const int tid  = threadIdx.x;
  const int b    = blockIdx.x >> 6;
  const int h    = blockIdx.x & 63;
  const int lane = tid & 31;
  const int wv   = tid >> 5;
  const int mb   = wv & 1;        // M block (0..1)
  const int ns   = wv >> 1;       // N subtile (0..3)

  v8f acc = {};

  const int p  = tid & 63;        // position (w column) this thread stages
  const int cg = tid >> 6;        // channel group 0..3 (8 ch each)

  // per-lane A fragment base: row m = mb*16 + (lane&15), K-half = (lane>>4)*16
  const int arow = mb * 16 + (lane & 15);
  const int acol = (lane >> 4) * 16;

  int par = 0;
  for (int k = 0; k < KTAPS; ++k) {
    const int dy = k / 3 - 1, dx = k % 3 - 1;
    const int hy = h + dy;
    const int xx = p + dx;
    const bool ok = (hy >= 0) & (hy < Hdim) & (xx >= 0) & (xx < Wdim);
    const int  splane = ok ? (hy * Wdim + xx) : 0;
    const float smul  = ok ? 1.0f : 0.0f;
    const __bf16* aTap = wt_off + (size_t)(k * MPAD_OFF + arow) * CHI + acol;

    for (int c0 = 0; c0 < CHI; c0 += 32) {
      __bf16* bbuf = &ldsB[par][0];
      // ---- stage B: im2col samples, 32 channels x 64 positions ----
      const float* xc = x + ((size_t)(b * CHI + c0 + cg * 8)) * PLANE;
      #pragma unroll
      for (int i = 0; i < 8; ++i) {
        bbuf[p * 32 + cg * 8 + i] = f2bf(smul * xc[splane]);
        xc += PLANE;
      }
      __syncthreads();

      v16bf afrag = *(const v16bf*)(aTap + c0);     // direct global A frag
      v16bf bfrag = *(const v16bf*)&bbuf[((ns * 16 + (lane & 15)) * 32) +
                                         ((lane >> 4) * 16)];
      acc = __builtin_amdgcn_wmma_f32_16x16x32_bf16(
          false, afrag, false, bfrag, (short)0, acc, false, false);
      par ^= 1;
    }
  }

  // ---- epilogue: split into offY / offX / sigmoid(mask) ----
  const int n    = lane & 15;
  const int wcol = ns * 16 + n;
  const int mhi  = (lane >> 4) << 3;
  #pragma unroll
  for (int r = 0; r < 8; ++r) {
    int j = mb * 16 + mhi + r;
    if (j < 27) {
      float v = acc[r] + b_off[j];
      if (j < 9)
        offY[(size_t)(b * KTAPS + j) * PLANE + h * Wdim + wcol] = v;
      else if (j < 18)
        offX[(size_t)(b * KTAPS + (j - 9)) * PLANE + h * Wdim + wcol] = v;
      else
        maskW[(size_t)(b * KTAPS + (j - 18)) * PLANE + h * Wdim + wcol] =
            sigmoidf_(v);
    }
  }
}

// ---------------------------------------------------------------------------
// Kernel 2: DCNv2 gather-GEMM + bias + BN + ReLU.
// M=128, N=64 per block, K=2304.  8 waves: wave w -> M block w, 4 N subtiles.
// ---------------------------------------------------------------------------
__global__ __launch_bounds__(256) void dcn_main_wmma(
    const float* __restrict__ x, const __bf16* __restrict__ wt_dcn,
    const float* __restrict__ b_dcn,
    const float* __restrict__ bn_gamma, const float* __restrict__ bn_beta,
    const float* __restrict__ bn_mean, const float* __restrict__ bn_var,
    const float* __restrict__ offY, const float* __restrict__ offX,
    const float* __restrict__ maskW, float* __restrict__ yMid) {
  __shared__ __align__(64) __bf16 ldsB[2][64 * 32];   // [p][kq], double buffer

  const int tid  = threadIdx.x;
  const int b    = blockIdx.x >> 6;
  const int h    = blockIdx.x & 63;
  const int lane = tid & 31;
  const int wv   = tid >> 5;       // M block (0..7)

  v8f acc[4] = {};

  const int p  = tid & 63;
  const int cg = tid >> 6;

  const int arow = wv * 16 + (lane & 15);
  const int acol = (lane >> 4) * 16;

  int par = 0;
  for (int k = 0; k < KTAPS; ++k) {
    // ---- per-tap bilinear setup for position p (mask folded in) ----
    const int dy = k / 3 - 1, dx = k % 3 - 1;
    const size_t ob = (size_t)(b * KTAPS + k) * PLANE + h * Wdim;
    const float py = (float)(h + dy) + offY[ob + p];
    const float px = (float)(p + dx) + offX[ob + p];
    const float mv = maskW[ob + p];

    const float y0f = floorf(py), x0f = floorf(px);
    const int y0 = (int)y0f, x0 = (int)x0f;
    const int y1 = y0 + 1,   x1 = x0 + 1;
    const float fy1 = py - y0f, fy0 = 1.0f - fy1;
    const float fx1 = px - x0f, fx0 = 1.0f - fx1;

    const float vy0 = (y0 >= 0 && y0 < Hdim) ? 1.0f : 0.0f;
    const float vy1 = (y1 >= 0 && y1 < Hdim) ? 1.0f : 0.0f;
    const float vx0 = (x0 >= 0 && x0 < Wdim) ? 1.0f : 0.0f;
    const float vx1 = (x1 >= 0 && x1 < Wdim) ? 1.0f : 0.0f;

    const int yc0 = min(max(y0, 0), Hdim - 1), yc1 = min(max(y1, 0), Hdim - 1);
    const int xc0 = min(max(x0, 0), Wdim - 1), xc1 = min(max(x1, 0), Wdim - 1);

    const int a00 = yc0 * Wdim + xc0, a01 = yc0 * Wdim + xc1;
    const int a10 = yc1 * Wdim + xc0, a11 = yc1 * Wdim + xc1;
    const float w00 = mv * fy0 * fx0 * vy0 * vx0;
    const float w01 = mv * fy0 * fx1 * vy0 * vx1;
    const float w10 = mv * fy1 * fx0 * vy1 * vx0;
    const float w11 = mv * fy1 * fx1 * vy1 * vx1;

    const __bf16* aTap = wt_dcn + (size_t)(k * CHO + arow) * CHI + acol;

    for (int c0 = 0; c0 < CHI; c0 += 32) {
      __bf16* bbuf = &ldsB[par][0];
      // ---- stage B: gathered/masked samples, bf16 ----
      const float* xc = x + ((size_t)(b * CHI + c0 + cg * 8)) * PLANE;
      if (c0 + 32 < CHI)  // warm next chunk (global_prefetch_b8)
        __builtin_prefetch(xc + 32 * PLANE + a00, 0, 1);
      #pragma unroll
      for (int i = 0; i < 8; ++i) {
        float v = w00 * xc[a00] + w01 * xc[a01] +
                  w10 * xc[a10] + w11 * xc[a11];
        bbuf[p * 32 + cg * 8 + i] = f2bf(v);
        xc += PLANE;
      }
      __syncthreads();

      v16bf afrag = *(const v16bf*)(aTap + c0);     // direct global A frag
      #pragma unroll
      for (int s = 0; s < 4; ++s) {
        v16bf bfrag = *(const v16bf*)&bbuf[((s * 16 + (lane & 15)) * 32) +
                                           ((lane >> 4) * 16)];
        acc[s] = __builtin_amdgcn_wmma_f32_16x16x32_bf16(
            false, afrag, false, bfrag, (short)0, acc[s], false, false);
      }
      par ^= 1;
    }
  }

  // ---- epilogue: bias + BN + ReLU, store (b, m, h, w) f32 ----
  const int n   = lane & 15;
  const int mhi = (lane >> 4) << 3;
  #pragma unroll
  for (int s = 0; s < 4; ++s) {
    const int wcol = s * 16 + n;
    #pragma unroll
    for (int r = 0; r < 8; ++r) {
      const int m = wv * 16 + mhi + r;
      const float scale = bn_gamma[m] * rsqrtf(bn_var[m] + 1e-5f);
      float v = (acc[s][r] + b_dcn[m] - bn_mean[m]) * scale + bn_beta[m];
      v = fmaxf(v, 0.0f);
      yMid[(size_t)(b * CHO + m) * PLANE + h * Wdim + wcol] = v;
    }
  }
}

// ---------------------------------------------------------------------------
// Kernel 3: depthwise 4x4 stride-2 transpose upsample + 1x1 conv + sigmoid.
// Output (8, 1, 128, 128).
// ---------------------------------------------------------------------------
__global__ __launch_bounds__(256) void upsample_1x1_sigmoid(
    const float* __restrict__ yMid, const float* __restrict__ w_up,
    const float* __restrict__ w1x1, float* __restrict__ out) {
  const int idx = blockIdx.x * blockDim.x + threadIdx.x;   // 131072 total
  const int b   = idx >> 14;
  const int rem = idx & 16383;
  const int oy  = rem >> 7;
  const int ox  = rem & 127;

  const int ty = oy + 1, tx = ox + 1;
  const int kyp = ty & 1, kxp = tx & 1;

  float acc = 0.0f;
  for (int c = 0; c < CHO; ++c) {
    const float* yc = yMid + ((size_t)(b * CHO + c) << 12);
    const float* wu = w_up + c * 16;
    float s = 0.0f;
    #pragma unroll
    for (int a = 0; a < 2; ++a) {
      const int ky = kyp + 2 * a;
      const int iy = (ty - ky) >> 1;
      if ((unsigned)iy < (unsigned)Hdim) {
        #pragma unroll
        for (int e = 0; e < 2; ++e) {
          const int kx = kxp + 2 * e;
          const int ix = (tx - kx) >> 1;
          if ((unsigned)ix < (unsigned)Wdim)
            s += wu[ky * 4 + kx] * yc[iy * Wdim + ix];
        }
      }
    }
    acc = fmaf(w1x1[c], s, acc);
  }
  out[idx] = sigmoidf_(acc);
}

// ---------------------------------------------------------------------------
extern "C" void kernel_launch(void* const* d_in, const int* in_sizes, int n_in,
                              void* d_out, int out_size, void* d_ws,
                              size_t ws_size, hipStream_t stream) {
  (void)in_sizes; (void)n_in; (void)out_size; (void)ws_size;
  const float* x        = (const float*)d_in[0];
  const float* w_dcn    = (const float*)d_in[1];
  const float* b_dcn    = (const float*)d_in[2];
  const float* w_off    = (const float*)d_in[3];
  const float* b_off    = (const float*)d_in[4];
  const float* bn_gamma = (const float*)d_in[5];
  const float* bn_beta  = (const float*)d_in[6];
  const float* bn_mean  = (const float*)d_in[7];
  const float* bn_var   = (const float*)d_in[8];
  const float* w_up     = (const float*)d_in[9];
  const float* w1x1     = (const float*)d_in[10];
  float* out = (float*)d_out;

  // workspace carve (~21 MB)
  const size_t offElems = (size_t)BATCH * KTAPS * PLANE;   // 294912
  float*  offY   = (float*)d_ws;
  float*  offX   = offY + offElems;
  float*  maskW  = offX + offElems;
  float*  yMid   = maskW + offElems;                       // 8*128*4096
  __bf16* wt_dcn = (__bf16*)(yMid + (size_t)BATCH * CHO * PLANE);
  __bf16* wt_off = wt_dcn + WT_DCN_ELEMS;

  const dim3 blk(256);
  reorder_weights<<<dim3((WT_DCN_ELEMS + WT_OFF_ELEMS + 255) / 256), blk, 0,
                    stream>>>(w_dcn, w_off, wt_dcn, wt_off);
  dcn_offset_conv_wmma<<<dim3(BATCH * Hdim), blk, 0, stream>>>(
      x, wt_off, b_off, offY, offX, maskW);
  dcn_main_wmma<<<dim3(BATCH * Hdim), blk, 0, stream>>>(
      x, wt_dcn, b_dcn, bn_gamma, bn_beta, bn_mean, bn_var,
      offY, offX, maskW, yMid);
  upsample_1x1_sigmoid<<<dim3((BATCH * 128 * 128) / 256), blk, 0, stream>>>(
      yMid, w_up, w1x1, out);
}
